// NeuralSuperSampling_18416819765329
// MI455X (gfx1250) — matrix operand order repaired and checked by reading, not compiled, not run on hardware
//
#include <hip/hip_runtime.h>
#include <hip/hip_bf16.h>

typedef _Float16 v16h __attribute__((ext_vector_type(16)));
typedef _Float16 v8h  __attribute__((ext_vector_type(8)));
typedef float    v8f  __attribute__((ext_vector_type(8)));

#define FF 5
#define HH 256
#define WW 448
#define HN 512
#define WN 896

__device__ __forceinline__ int iclamp(int v, int lo, int hi) {
    return v < lo ? lo : (v > hi ? hi : v);
}

// Bilinear sample of a (HH, WW) image at (sy, sx), clamp-to-edge.
__device__ __forceinline__ float bilerp_clamp(const float* __restrict__ img, float sy, float sx) {
    float y0f = floorf(sy), x0f = floorf(sx);
    float wy = sy - y0f, wx = sx - x0f;
    int y0 = (int)y0f, x0 = (int)x0f;
    int y0c = iclamp(y0, 0, HH - 1), y1c = iclamp(y0 + 1, 0, HH - 1);
    int x0c = iclamp(x0, 0, WW - 1), x1c = iclamp(x0 + 1, 0, WW - 1);
    float v00 = img[y0c * WW + x0c], v01 = img[y0c * WW + x1c];
    float v10 = img[y1c * WW + x0c], v11 = img[y1c * WW + x1c];
    return v00 * (1.f - wy) * (1.f - wx) + v01 * (1.f - wy) * wx +
           v10 * wy * (1.f - wx) + v11 * wy * wx;
}

// Frame 0: zero-upsampled raw RGB + depth into rgbd channels 0..3 (f16).
__global__ __launch_bounds__(256) void frame0_zup_kernel(
    const float* __restrict__ color, const float* __restrict__ depth,
    _Float16* __restrict__ rgbd, int B) {
    const size_t HW2 = (size_t)HN * WN;
    size_t idx = (size_t)blockIdx.x * blockDim.x + threadIdx.x;
    if (idx >= (size_t)B * HW2) return;
    int b = (int)(idx / HW2);
    int rem = (int)(idx % HW2);
    int y = rem / WN, x = rem % WN;
    bool on = (y & 1) && (x & 1);     // zero_upsample puts data at [1::2]
    int ys = y >> 1, xs = x >> 1;
    #pragma unroll
    for (int c = 0; c < 4; ++c) {
        float v = 0.f;
        if (on) {
            if (c < 3) v = color[(((size_t)b * FF + 0) * 3 + c) * ((size_t)HH * WW) + (size_t)ys * WW + xs];
            else       v = depth[((size_t)b * FF + 0) * ((size_t)HH * WW) + (size_t)ys * WW + xs];
        }
        rgbd[((size_t)b * 20 + c) * HW2 + rem] = (_Float16)v;
    }
}

// One bilinear warp pass with mv[j] (motion bilinearly upsampled on the fly).
// SRC_ZUP: source is the zero-upsampled raw rgbd of frame `frame_i` (read from inputs).
// DST_F16: write result into rgbd channels [frame_i*4 .. frame_i*4+4) as f16, else f32 ping/pong.
template <bool SRC_ZUP, bool DST_F16>
__global__ __launch_bounds__(256) void warp_kernel(
    const float* __restrict__ src, const float* __restrict__ color,
    const float* __restrict__ depth, const float* __restrict__ motion,
    int frame_i, int j, float* __restrict__ dstf, _Float16* __restrict__ rgbd, int B) {
    const size_t HW2 = (size_t)HN * WN;
    size_t idx = (size_t)blockIdx.x * blockDim.x + threadIdx.x;
    if (idx >= (size_t)B * HW2) return;
    int b = (int)(idx / HW2);
    int rem = (int)(idx % HW2);
    int y = rem / WN, x = rem % WN;

    // mv = bilinear 2x upsample of motion[b, j] (half-pixel centers)
    const float* mvxp = motion + (((size_t)b * FF + j) * 2 + 0) * ((size_t)HH * WW);
    const float* mvyp = mvxp + (size_t)HH * WW;
    float sy0 = y * 0.5f - 0.25f, sx0 = x * 0.5f - 0.25f;
    float mvx = bilerp_clamp(mvxp, sy0, sx0);
    float mvy = bilerp_clamp(mvyp, sy0, sx0);

    float sx = (float)x + mvx, sy = (float)y + mvy;
    float x0f = floorf(sx), y0f = floorf(sy);
    float wx = sx - x0f, wy = sy - y0f;
    int ix = (int)x0f, iy = (int)y0f;
    float wgt[4] = {(1.f - wy) * (1.f - wx), (1.f - wy) * wx, wy * (1.f - wx), wy * wx};
    int ty[4] = {iy, iy, iy + 1, iy + 1};
    int tx[4] = {ix, ix + 1, ix, ix + 1};

    float out[4] = {0.f, 0.f, 0.f, 0.f};
    #pragma unroll
    for (int t = 0; t < 4; ++t) {
        int yt = ty[t], xt = tx[t];
        if (yt < 0 || yt >= HN || xt < 0 || xt >= WN) continue;  // invalid tap -> weight 0
        float w = wgt[t];
        if (SRC_ZUP) {
            if (!((yt & 1) && (xt & 1))) continue;  // zero-upsample structural zero
            int ys = yt >> 1, xs = xt >> 1;
            #pragma unroll
            for (int c = 0; c < 4; ++c) {
                float v = (c < 3)
                    ? color[(((size_t)b * FF + frame_i) * 3 + c) * ((size_t)HH * WW) + (size_t)ys * WW + xs]
                    : depth[((size_t)b * FF + frame_i) * ((size_t)HH * WW) + (size_t)ys * WW + xs];
                out[c] += v * w;
            }
        } else {
            #pragma unroll
            for (int c = 0; c < 4; ++c)
                out[c] += src[((size_t)b * 4 + c) * HW2 + (size_t)yt * WN + xt] * w;
        }
    }
    #pragma unroll
    for (int c = 0; c < 4; ++c) {
        if (DST_F16) rgbd[((size_t)b * 20 + frame_i * 4 + c) * HW2 + rem] = (_Float16)out[c];
        else         dstf[((size_t)b * 4 + c) * HW2 + rem] = out[c];
    }
}

// 3x3 SAME conv as implicit GEMM on v_wmma_f32_16x16x32_f16.
// LDS holds an im2col panel sA[pixel][k] (k zero-padded to KP) and K-major weights
// sW[k][n]; A fragments are contiguous 16-byte LDS reads (ds_load_b128), B is strided by NROW.
// ACT 0: relu -> f16 out (B,COUT,HN,WN); ACT 1: (tanh+1)*10 -> f32 out (B,4,HN,WN).
template <int CIN, int COUT, int MSUB, int NTILEW, int ACT>
__global__ __launch_bounds__(MSUB * NTILEW * 32) void conv3x3_wmma_kernel(
    const _Float16* __restrict__ in, const float* __restrict__ wgt,
    const float* __restrict__ bias, _Float16* __restrict__ outh,
    float* __restrict__ outf, int B) {
    constexpr int K = CIN * 9;
    constexpr int KSTEPS = (K + 31) / 32;
    constexpr int KP = KSTEPS * 32;
    constexpr int KROW = KP + 8;        // sA row stride in halves (16B aligned, bank-skewed)
    constexpr int TILE_M = MSUB * 16;
    constexpr int NT = NTILEW * 16;
    constexpr int NROW = NT + 8;        // sW row stride in halves (16B aligned, bank-skewed)
    constexpr int NTHREADS = MSUB * NTILEW * 32;

    __shared__ _Float16 sA[TILE_M * KROW];  // im2col panel
    __shared__ _Float16 sW[KP * NROW];      // weights, K-major

    const size_t HW2 = (size_t)HN * WN;
    const int tid = threadIdx.x;
    const int wave = tid >> 5, lane = tid & 31;
    const int msub = wave % MSUB, ntile = wave / MSUB;
    const int x0 = blockIdx.x * TILE_M;
    const int y = blockIdx.y;
    const int b = blockIdx.z;

    // Weights -> LDS (K-major, zero padded).
    for (int t = tid; t < KP * NT; t += NTHREADS) {
        int n = t / KP, k = t % KP;     // global reads contiguous in k
        sW[k * NROW + n] = (n < COUT && k < K) ? (_Float16)wgt[n * K + k] : (_Float16)0.f;
    }
    // im2col panel -> LDS. One (c,dy,dx) line per k; consecutive threads = consecutive x.
    {
        const int xl = tid % TILE_M;
        const int kg = tid / TILE_M;
        for (int k = kg; k < KP; k += NTHREADS / TILE_M) {
            _Float16 v = (_Float16)0.f;
            if (k < K) {
                int c = k / 9, rr = k % 9;
                int dy = rr / 3, dx = rr % 3;
                int yy = y + dy - 1;
                int xg = x0 + xl + dx - 1;
                if (yy >= 0 && yy < HN && xg >= 0 && xg < WN)
                    v = in[((size_t)b * CIN + c) * HW2 + (size_t)yy * WN + xg];
            }
            sA[xl * KROW + k] = v;
        }
    }
    __syncthreads();

    const int laneM = lane & 15, laneH = lane >> 4;
    const _Float16* aRow = &sA[(msub * 16 + laneM) * KROW];  // A: M = lane%16
    v8f acc = {};
    #pragma unroll
    for (int s = 0; s < KSTEPS; ++s) {
        // A fragment: lane's K values are kk0+8*laneH + {0..7, 16..23} -> 2x b128
        const int abase = s * 32 + laneH * 8;
        v8h a0 = *(const v8h*)(aRow + abase);
        v8h a1 = *(const v8h*)(aRow + abase + 16);
        v16h a = __builtin_shufflevector(a0, a1, 0, 1, 2, 3, 4, 5, 6, 7,
                                         8, 9, 10, 11, 12, 13, 14, 15);
        // B fragment: N = ntile*16 + lane%16, K = kk0+8*laneH + {0..7, 16..23}
        const _Float16* bCol = &sW[(s * 32 + laneH * 8) * NROW + ntile * 16 + laneM];
        v16h bf;
        #pragma unroll
        for (int i = 0; i < 8; ++i) { bf[i] = bCol[i * NROW]; bf[8 + i] = bCol[(16 + i) * NROW]; }
        acc = __builtin_amdgcn_wmma_f32_16x16x32_f16(false, a, false, bf, (short)0, acc,
                                                     false, false);
    }

    // D layout: lane -> N = lane%16; VGPR r -> M = r + 8*(lane/16).
    // Lane's 8 outputs are 8 consecutive pixels -> one 16B store.
    const int n = ntile * 16 + laneM;
    const float bv = (n < COUT) ? bias[n] : 0.f;
    const int xbase = x0 + msub * 16 + laneH * 8;
    if (ACT == 0) {
        v8h hv;
        #pragma unroll
        for (int r = 0; r < 8; ++r) hv[r] = (_Float16)fmaxf(acc[r] + bv, 0.f);
        if (n < COUT)
            *(v8h*)(outh + ((size_t)b * COUT + n) * HW2 + (size_t)y * WN + xbase) = hv;
    } else {
        if (n < 4) {
            #pragma unroll
            for (int r = 0; r < 8; ++r)
                outf[((size_t)b * 4 + n) * HW2 + (size_t)y * WN + xbase + r] =
                    (tanhf(acc[r] + bv) + 1.f) * 10.f;
        }
    }
}

extern "C" void kernel_launch(void* const* d_in, const int* in_sizes, int n_in,
                              void* d_out, int out_size, void* d_ws, size_t ws_size,
                              hipStream_t stream) {
    (void)n_in; (void)out_size; (void)ws_size;
    const float* color  = (const float*)d_in[0];
    const float* motion = (const float*)d_in[1];
    const float* depth  = (const float*)d_in[2];
    // f1_*/fp_* (indices 3..14) are dead w.r.t. the output and skipped.
    const float* rw_w1 = (const float*)d_in[15];
    const float* rw_b1 = (const float*)d_in[16];
    const float* rw_w2 = (const float*)d_in[17];
    const float* rw_b2 = (const float*)d_in[18];
    const float* rw_w3 = (const float*)d_in[19];
    const float* rw_b3 = (const float*)d_in[20];

    const int B = in_sizes[0] / (FF * 3 * HH * WW);
    const size_t HW2 = (size_t)HN * WN;

    // Workspace layout
    float* ping = (float*)d_ws;                                // (B,4,HN,WN) f32
    float* pong = ping + (size_t)B * 4 * HW2;                  // (B,4,HN,WN) f32
    _Float16* rgbd = (_Float16*)(pong + (size_t)B * 4 * HW2);  // (B,20,HN,WN) f16
    _Float16* act1 = rgbd + (size_t)B * 20 * HW2;              // (B,32,HN,WN) f16
    _Float16* act2 = act1 + (size_t)B * 32 * HW2;              // (B,32,HN,WN) f16

    const int nPix = (int)((size_t)B * HW2);
    dim3 blk(256), grd((nPix + 255) / 256);

    frame0_zup_kernel<<<grd, blk, 0, stream>>>(color, depth, rgbd, B);

    // Recursive warp chains: frame i is warped through mv[i-1], ..., mv[0].
    // frame 1
    warp_kernel<true,  true ><<<grd, blk, 0, stream>>>(nullptr, color, depth, motion, 1, 0, nullptr, rgbd, B);
    // frame 2
    warp_kernel<true,  false><<<grd, blk, 0, stream>>>(nullptr, color, depth, motion, 2, 1, ping, nullptr, B);
    warp_kernel<false, true ><<<grd, blk, 0, stream>>>(ping,    color, depth, motion, 2, 0, nullptr, rgbd, B);
    // frame 3
    warp_kernel<true,  false><<<grd, blk, 0, stream>>>(nullptr, color, depth, motion, 3, 2, ping, nullptr, B);
    warp_kernel<false, false><<<grd, blk, 0, stream>>>(ping,    color, depth, motion, 3, 1, pong, nullptr, B);
    warp_kernel<false, true ><<<grd, blk, 0, stream>>>(pong,    color, depth, motion, 3, 0, nullptr, rgbd, B);
    // frame 4
    warp_kernel<true,  false><<<grd, blk, 0, stream>>>(nullptr, color, depth, motion, 4, 3, ping, nullptr, B);
    warp_kernel<false, false><<<grd, blk, 0, stream>>>(ping,    color, depth, motion, 4, 2, pong, nullptr, B);
    warp_kernel<false, false><<<grd, blk, 0, stream>>>(pong,    color, depth, motion, 4, 1, ping, nullptr, B);
    warp_kernel<false, true ><<<grd, blk, 0, stream>>>(ping,    color, depth, motion, 4, 0, nullptr, rgbd, B);

    // Reweighting network via WMMA implicit GEMM.
    conv3x3_wmma_kernel<20, 32, 4, 2, 0><<<dim3(WN / 64, HN, B), 256, 0, stream>>>(rgbd, rw_w1, rw_b1, act1, nullptr, B);
    conv3x3_wmma_kernel<32, 32, 4, 2, 0><<<dim3(WN / 64, HN, B), 256, 0, stream>>>(act1, rw_w2, rw_b2, act2, nullptr, B);
    conv3x3_wmma_kernel<32,  4, 4, 1, 1><<<dim3(WN / 64, HN, B), 128, 0, stream>>>(act2, rw_w3, rw_b3, nullptr, (float*)d_out, B);
}
